// TinyMultiHeadAttention_48936857370779
// MI455X (gfx1250) — matrix operand order repaired and checked by reading, not compiled, not run on hardware
//
#include <hip/hip_runtime.h>
#include <cstdint>

// ---------------------------------------------------------------------------
// CDNA5 (gfx1250, wave32) multi-head attention.
// bf16 WMMA + fp32 accumulate; async global->LDS DMA, double-buffered.
// ---------------------------------------------------------------------------

typedef __attribute__((ext_vector_type(16))) __bf16 v16bf;
typedef __attribute__((ext_vector_type(8)))  __bf16 v8bf;
typedef __attribute__((ext_vector_type(8)))  float  v8f;

#define WMMA_BF16(a, b, c)                                                    \
  __builtin_amdgcn_wmma_f32_16x16x32_bf16(false, (a), false, (b), (short)0,   \
                                          (c), false, false)
#define EXP2(x) __builtin_amdgcn_exp2f(x)  // raw v_exp_f32, no denorm fixup

// Async 16-byte global -> LDS copy (ASYNCcnt-tracked, no VGPR round trip).
__device__ __forceinline__ void async_copy16(uint32_t lds_byte,
                                             const void* gptr) {
  asm volatile("global_load_async_to_lds_b128 %0, %1, off"
               :
               : "v"(lds_byte), "v"((uint64_t)(uintptr_t)gptr)
               : "memory");
}
__device__ __forceinline__ void wait_async0() {
  asm volatile("s_wait_asynccnt 0" ::: "memory");
}

// 16 contiguous bf16 -> fragment (B-fragments stored k-contiguous).
__device__ __forceinline__ v16bf load_contig16(const __bf16* p) {
  v8bf lo = *reinterpret_cast<const v8bf*>(p);
  v8bf hi = *reinterpret_cast<const v8bf*>(p + 8);
  v16bf r;
#pragma unroll
  for (int e = 0; e < 8; ++e) { r[e] = lo[e]; r[e + 8] = hi[e]; }
  return r;
}

// A-matrix fragment: 16x32 bf16 tile, row-major, leading dim ld.
__device__ __forceinline__ v16bf load_a_frag(const __bf16* base, int ld,
                                             int lane) {
  const __bf16* p = base + (size_t)(lane & 15) * ld + ((lane >> 4) << 4);
  v8bf lo = *reinterpret_cast<const v8bf*>(p);
  v8bf hi = *reinterpret_cast<const v8bf*>(p + 8);
  v16bf r;
#pragma unroll
  for (int e = 0; e < 8; ++e) { r[e] = lo[e]; r[e + 8] = hi[e]; }
  return r;
}

// ---------------------------------------------------------------------------
// One-time preprocessing: fp32 -> bf16 activations; weights -> bf16 [n][k].
// ---------------------------------------------------------------------------
__global__ __launch_bounds__(256) void cvt_bf16(const float* __restrict__ in,
                                                __bf16* __restrict__ out,
                                                int n8) {
  const int i = blockIdx.x * 256 + threadIdx.x;
  if (i >= n8) return;
  const float4* p = (const float4*)in + (size_t)i * 2;
  const float4 a = p[0], b = p[1];
  v8bf o;
  o[0] = (__bf16)a.x; o[1] = (__bf16)a.y; o[2] = (__bf16)a.z; o[3] = (__bf16)a.w;
  o[4] = (__bf16)b.x; o[5] = (__bf16)b.y; o[6] = (__bf16)b.z; o[7] = (__bf16)b.w;
  *((v8bf*)out + i) = o;
}

__global__ __launch_bounds__(256) void transpose_cvt_bf16(
    const float* __restrict__ W, __bf16* __restrict__ WT, int N, int K) {
  const int idx = blockIdx.x * 256 + threadIdx.x;  // N*K/8 threads
  const int n = idx >> 7;                          // K/8 == 128
  const int k8 = (idx & 127) << 3;
  v8bf o;
#pragma unroll
  for (int j = 0; j < 8; ++j) o[j] = (__bf16)W[(size_t)(k8 + j) * N + n];
  *(v8bf*)(WT + (size_t)n * K + k8) = o;
}

// ---------------------------------------------------------------------------
// GEMM: C[M,N] = (A[M,K] @ WT[N,K]^T + bias) * out_scale, bf16 WMMA.
// A bf16 [M][K]; WT bf16 [N][K]. Inner loop: double-buffered async staging.
// OUT_LAYOUT: 0 = fp32 [M][N]; 1 = bf16 [B,H,S,64]; 2 = bf16 [B,H,64,S].
// ---------------------------------------------------------------------------
template <int OUT_LAYOUT>
__global__ __launch_bounds__(256) void gemm_bias_wmma(
    const __bf16* __restrict__ A, const __bf16* __restrict__ WT,
    const float* __restrict__ bias, void* __restrict__ outv, int M, int N,
    int K, float out_scale) {
  __shared__ __bf16 As[2][64][32];  // [m][k], 8 KB
  __shared__ __bf16 Ws[2][64][32];  // [n][k], 8 KB
  const int tid = threadIdx.x;
  const int lane = tid & 31;
  const int wave = tid >> 5;
  const int m0 = blockIdx.y * 64;
  const int n0 = blockIdx.x * 64;
  const int wm = wave >> 1;        // 0..3 : 16-row slice
  const int wn = (wave & 1) << 1;  // 0/2  : two 16-col slices per wave

  const uint32_t ldsA = (uint32_t)(uintptr_t)(&As[0][0][0]);
  const uint32_t ldsW = (uint32_t)(uintptr_t)(&Ws[0][0][0]);
  const int srow = tid >> 2;       // 0..63: staged row
  const int scb = (tid & 3) * 16;  // byte chunk within 64-byte row

  v8f c0 = {};
  v8f c1 = {};
  // Issue first K-slab (k0 = 0).
  async_copy16(ldsA + srow * 64u + scb,
               (const char*)(A + (size_t)(m0 + srow) * K) + scb);
  async_copy16(ldsW + srow * 64u + scb,
               (const char*)(WT + (size_t)(n0 + srow) * K) + scb);

  for (int k0 = 0; k0 < K; k0 += 32) {
    const int buf = (k0 >> 5) & 1;
    wait_async0();
    __syncthreads();    // slab `buf` resident for all waves
    if (k0 + 32 < K) {  // prefetch next slab into other buffer
      const uint32_t lo = (buf ^ 1) * 4096u + srow * 64u + scb;
      async_copy16(ldsA + lo,
                   (const char*)(A + (size_t)(m0 + srow) * K + k0 + 32) + scb);
      async_copy16(ldsW + lo,
                   (const char*)(WT + (size_t)(n0 + srow) * K + k0 + 32) + scb);
    }
    v16bf a = load_a_frag(&As[buf][wm << 4][0], 32, lane);
    v16bf b0 = load_contig16(
        &Ws[buf][((wn + 0) << 4) + (lane & 15)][(lane >> 4) << 4]);
    v16bf b1 = load_contig16(
        &Ws[buf][((wn + 1) << 4) + (lane & 15)][(lane >> 4) << 4]);
    c0 = WMMA_BF16(a, b0, c0);
    c1 = WMMA_BF16(a, b1, c1);
  }

  // Epilogue: C layout -> lane L holds rows r+8*(L/16), col = L%16.
  const int rowbase = m0 + (wm << 4) + ((lane >> 4) << 3);
  const int ncol0 = n0 + (wn << 4) + (lane & 15);
#pragma unroll
  for (int t = 0; t < 2; ++t) {
    const v8f& c = t ? c1 : c0;
    const int ng = ncol0 + (t << 4);
    const float bv = bias[ng];
    if (OUT_LAYOUT == 0) {
      float* o = (float*)outv;
#pragma unroll
      for (int r = 0; r < 8; ++r)
        o[(size_t)(rowbase + r) * N + ng] = (c[r] + bv) * out_scale;
    } else if (OUT_LAYOUT == 1) {  // bf16 [b, h, s, d]
      __bf16* o = (__bf16*)outv;
      const int h = ng >> 6, d = ng & 63;
#pragma unroll
      for (int r = 0; r < 8; ++r) {
        const int mg = rowbase + r;
        o[((((size_t)(mg >> 10) * 16 + h) * 1024) + (mg & 1023)) * 64 + d] =
            (__bf16)((c[r] + bv) * out_scale);
      }
    } else {  // bf16 [b, h, d, s]  (8 consecutive s per lane -> vector store)
      __bf16* o = (__bf16*)outv;
      const int h = ng >> 6, d = ng & 63;
      const int b = rowbase >> 10, s0 = rowbase & 1023;
      __bf16* orow = o + (((size_t)b * 16 + h) * 64 + d) * 1024 + s0;
#pragma unroll
      for (int r = 0; r < 8; ++r) orow[r] = (__bf16)((c[r] + bv) * out_scale);
    }
  }
}

// QK^T 16x16 logit tile (1/sqrt(dh) * log2(e) pre-folded into Q projection).
__device__ __forceinline__ v8f qk_tile(const __bf16* Kt, v16bf qa0, v16bf qa1,
                                       int lane) {
  const size_t roff = (size_t)(lane & 15) * 64 + ((lane >> 4) << 4);
  v16bf kb0 = load_contig16(Kt + roff);       // d =  0..31 slice
  v16bf kb1 = load_contig16(Kt + roff + 32);  // d = 32..63 slice
  v8f c = {};
  c = WMMA_BF16(qa0, kb0, c);
  c = WMMA_BF16(qa1, kb1, c);
  return c;
}

// ---------------------------------------------------------------------------
// Attention: one block = (b,h, 64 q-rows); 4 waves x 16 q-rows each.
// Pass 1: per-lane online softmax (base-2), stats pipelined one tile behind
//         the WMMA stream so VALU co-executes with the matrix pipe.
// Pass 2: recompute logits, write normalized attn once, attn@V via WMMA with
//         double-buffered async-DMA'd V^T tiles.
// ---------------------------------------------------------------------------
__global__ __launch_bounds__(128) void flash_attn_wmma(
    const __bf16* __restrict__ Qh, const __bf16* __restrict__ Kh,
    const __bf16* __restrict__ VhT, float* __restrict__ attn_out,
    __bf16* __restrict__ ctxb) {
  const int S = 1024, DH = 64;
  __shared__ __bf16 Vs[2][64][32];  // V^T tile [d][key], double-buffered, 8 KB
  __shared__ __bf16 Ps[4][16][32];  // per-wave softmax tile (A for attn@V)
  const int tid = threadIdx.x, lane = tid & 31, wave = tid >> 5;
  const int qt = blockIdx.x & 15;  // S/64 = 16 q-tiles
  const int bh = blockIdx.x >> 4;  // b*H + h
  const int q0 = qt * 64 + wave * 16;
  const __bf16* Qb = Qh + ((size_t)bh * S + q0) * DH;
  const __bf16* Kb = Kh + (size_t)bh * S * DH;
  const __bf16* Vb = VhT + (size_t)bh * DH * S;  // [d][s]

  v16bf qa0 = load_a_frag(Qb, DH, lane);
  v16bf qa1 = load_a_frag(Qb + 32, DH, lane);

  float m[8], l[8];
#pragma unroll
  for (int r = 0; r < 8; ++r) { m[r] = -3.0e38f; l[r] = 0.f; }

  // ---- Pass 1: per-lane online (max, sum-exp2), pipelined one tile ----
  v8f cp = qk_tile(Kb, qa0, qa1, lane);
  for (int kt = 1; kt < S / 16; ++kt) {
    v8f cn = qk_tile(Kb + (size_t)kt * 16 * DH, qa0, qa1, lane);  // overlap
#pragma unroll
    for (int r = 0; r < 8; ++r) {
      const float v = cp[r];
      const float mn = fmaxf(m[r], v);
      l[r] = l[r] * EXP2(m[r] - mn) + EXP2(v - mn);
      m[r] = mn;
    }
    cp = cn;
  }
#pragma unroll
  for (int r = 0; r < 8; ++r) {  // drain last tile
    const float v = cp[r];
    const float mn = fmaxf(m[r], v);
    l[r] = l[r] * EXP2(m[r] - mn) + EXP2(v - mn);
    m[r] = mn;
  }
#pragma unroll
  for (int r = 0; r < 8; ++r) {  // merge 16 lanes per half-wave
#pragma unroll
    for (int s = 1; s < 16; s <<= 1) {
      const float mo = __shfl_xor(m[r], s, 16);
      const float lo = __shfl_xor(l[r], s, 16);
      const float mn = fmaxf(m[r], mo);
      l[r] = l[r] * EXP2(m[r] - mn) + lo * EXP2(mo - mn);
      m[r] = mn;
    }
  }
  float il[8];
#pragma unroll
  for (int r = 0; r < 8; ++r) il[r] = __builtin_amdgcn_rcpf(l[r]);  // l >= 1

  // ---- Pass 2: attn out + ctx = attn @ V ----
  v8f acc[4] = {{}, {}, {}, {}};
  float* attn_base = attn_out + ((size_t)bh * S + q0) * S;
  const uint32_t ldsV = (uint32_t)(uintptr_t)(&Vs[0][0][0]);

  // V^T tile DMA: 64 d-rows x 32 keys (64 B), 2 chunks of 16 B per thread.
#define ISSUE_V(bufi, kk2)                                                    \
  {                                                                           \
    _Pragma("unroll") for (int cph = 0; cph < 2; ++cph) {                     \
      const int id = tid + cph * 128;                                         \
      const int row = id >> 2;                                                \
      const int cb = (id & 3) * 16;                                           \
      async_copy16(ldsV + (bufi) * 4096u + row * 64u + cb,                    \
                   (const char*)(Vb + (size_t)row * S + (kk2) * 32) + cb);    \
    }                                                                         \
  }

  ISSUE_V(0, 0);
  for (int kt2 = 0; kt2 < S / 32; ++kt2) {
    const int buf = kt2 & 1;
    wait_async0();
    __syncthreads();  // V tile `buf` resident for all waves
    if (kt2 + 1 < S / 32) ISSUE_V(buf ^ 1, kt2 + 1);
    // Compute both 16-key logit tiles first, then the exp/store work, so the
    // VALU stream does not immediately consume fresh WMMA outputs.
    v8f ca = qk_tile(Kb + (size_t)(kt2 * 2 + 0) * 16 * DH, qa0, qa1, lane);
    v8f cb = qk_tile(Kb + (size_t)(kt2 * 2 + 1) * 16 * DH, qa0, qa1, lane);
    const int col = lane & 15;
    const int rhi = (lane >> 4) << 3;
#pragma unroll
    for (int sub = 0; sub < 2; ++sub) {
      const v8f& c = sub ? cb : ca;
      const int kt = kt2 * 2 + sub;
#pragma unroll
      for (int r = 0; r < 8; ++r) {
        const float p = EXP2(c[r] - m[r]) * il[r];
        attn_base[(size_t)(rhi + r) * S + kt * 16 + col] = p;  // attn output
        Ps[wave][rhi + r][(sub << 4) + col] = (__bf16)p;
      }
    }
    __syncthreads();  // Ps cross-lane visibility
    v16bf pa = load_a_frag(&Ps[wave][0][0], 32, lane);
#pragma unroll
    for (int nb = 0; nb < 4; ++nb) {
      v16bf vb =
          load_contig16(&Vs[buf][(nb << 4) + (lane & 15)][(lane >> 4) << 4]);
      acc[nb] = WMMA_BF16(pa, vb, acc[nb]);
    }
  }
#undef ISSUE_V

  // ctx (bf16) [B,S,D], merging heads: col = h*64 + d.
  const int b = bh >> 4, h = bh & 15;
#pragma unroll
  for (int nb = 0; nb < 4; ++nb)
#pragma unroll
    for (int r = 0; r < 8; ++r) {
      const int row = ((lane >> 4) << 3) + r;
      ctxb[(size_t)(b * 1024 + q0 + row) * 1024 + h * 64 + (nb << 4) +
           (lane & 15)] = (__bf16)acc[nb][r];
    }
}

// ---------------------------------------------------------------------------
extern "C" void kernel_launch(void* const* d_in, const int* in_sizes, int n_in,
                              void* d_out, int out_size, void* d_ws,
                              size_t ws_size, hipStream_t stream) {
  (void)in_sizes; (void)n_in; (void)out_size; (void)ws_size;
  const float* query = (const float*)d_in[0];
  const float* key   = (const float*)d_in[1];
  const float* value = (const float*)d_in[2];
  const float* Wq = (const float*)d_in[3];
  const float* bq = (const float*)d_in[4];
  const float* Wk = (const float*)d_in[5];
  const float* bk = (const float*)d_in[6];
  const float* Wv = (const float*)d_in[7];
  const float* bv = (const float*)d_in[8];
  const float* Wo = (const float*)d_in[9];
  const float* bo = (const float*)d_in[10];

  const int B = 8, S = 1024, D = 1024, H = 16;
  const int M = B * S;  // 8192
  const size_t MB = 1024 * 1024;

  char* ws = (char*)d_ws;
  __bf16* Qh   = (__bf16*)(ws);             // bf16 [B,H,S,64]
  __bf16* Kh   = (__bf16*)(ws + 16 * MB);   // bf16 [B,H,S,64]
  __bf16* VhT  = (__bf16*)(ws + 32 * MB);   // bf16 [B,H,64,S]
  __bf16* ctxb = (__bf16*)(ws + 48 * MB);   // bf16 [B,S,D]
  __bf16* Aq   = (__bf16*)(ws + 64 * MB);   // bf16 activations
  __bf16* Ak   = (__bf16*)(ws + 80 * MB);
  __bf16* Av   = (__bf16*)(ws + 96 * MB);
  __bf16* WTq  = (__bf16*)(ws + 112 * MB);  // bf16 weights [N][K]
  __bf16* WTk  = (__bf16*)(ws + 114 * MB);
  __bf16* WTv  = (__bf16*)(ws + 116 * MB);
  __bf16* WTo  = (__bf16*)(ws + 118 * MB);

  float* outp = (float*)d_out;          // [B,S,D]
  float* attnp = outp + (size_t)M * D;  // [B,H,S,S]

  // 1/sqrt(64) * log2(e): softmax computed in base 2, scale folded into Q.
  const float qscale = 0.125f * 1.44269504088896340736f;

  // Preprocess: weights -> bf16 [n][k]; activations -> bf16.
  transpose_cvt_bf16<<<dim3((D * D / 8) / 256), dim3(256), 0, stream>>>(Wq, WTq, D, D);
  transpose_cvt_bf16<<<dim3((D * D / 8) / 256), dim3(256), 0, stream>>>(Wk, WTk, D, D);
  transpose_cvt_bf16<<<dim3((D * D / 8) / 256), dim3(256), 0, stream>>>(Wv, WTv, D, D);
  transpose_cvt_bf16<<<dim3((D * D / 8) / 256), dim3(256), 0, stream>>>(Wo, WTo, D, D);
  const int n8 = M * D / 8;
  cvt_bf16<<<dim3(n8 / 256), dim3(256), 0, stream>>>(query, Aq, n8);
  cvt_bf16<<<dim3(n8 / 256), dim3(256), 0, stream>>>(key, Ak, n8);
  cvt_bf16<<<dim3(n8 / 256), dim3(256), 0, stream>>>(value, Av, n8);

  dim3 gblock(256), ggrid(D / 64, M / 64);  // 16 x 128 tiles
  gemm_bias_wmma<1><<<ggrid, gblock, 0, stream>>>(Aq, WTq, bq, (void*)Qh, M, D,
                                                  D, qscale);
  gemm_bias_wmma<1><<<ggrid, gblock, 0, stream>>>(Ak, WTk, bk, (void*)Kh, M, D,
                                                  D, 1.0f);
  gemm_bias_wmma<2><<<ggrid, gblock, 0, stream>>>(Av, WTv, bv, (void*)VhT, M,
                                                  D, D, 1.0f);

  dim3 ablock(128), agrid(B * H * (S / 64));  // 2048 blocks
  flash_attn_wmma<<<agrid, ablock, 0, stream>>>(Qh, Kh, VhT, attnp, ctxb);

  gemm_bias_wmma<0><<<ggrid, gblock, 0, stream>>>(ctxb, WTo, bo, (void*)outp,
                                                  M, D, D, 1.0f);
}
